// RPE_40492951666721
// MI455X (gfx1250) — matrix operand order repaired and checked by reading, not compiled, not run on hardware
//
#include <hip/hip_runtime.h>
#include <hip/hip_bf16.h>

// Shapes from the reference
#define B_  2
#define D_  256
#define H_  8
#define T_  128
#define S_  128
#define F_  64
#define C_  512   // H*F
#define TE_ 512

typedef __attribute__((ext_vector_type(8)))  _Float16 v8h;
typedef __attribute__((ext_vector_type(16))) _Float16 v16h;
typedef __attribute__((ext_vector_type(8)))  float    v8f;
typedef __attribute__((ext_vector_type(4)))  float    v4f;

union V16 { v16h v; v8h h[2]; };

__device__ __forceinline__ v8f wmma_f16(v16h a, v16h b, v8f c) {
    // (neg_a, A, neg_b, B, c_mod, C, reuse_a, reuse_b)
    return __builtin_amdgcn_wmma_f32_16x16x32_f16(false, a, false, b, (short)0, c,
                                                  false, false);
}

// ---------------------------------------------------------------------------
// Kernel 1: proj[bt, c] = temb[bt, :] @ w_time[:, c] + b_time[c]
// 8 rows per block so w_time (1 MB) is streamed only 32x (32 MB of L2 reads)
// instead of 256x. Each thread: 2 channels x 8 rows = 16 accumulators.
// ---------------------------------------------------------------------------
#define RPB_ 8
__global__ __launch_bounds__(256) void temb_proj_kernel(
    const float* __restrict__ temb,    // (B*T, TE)
    const float* __restrict__ w_time,  // (TE, C) row-major
    const float* __restrict__ b_time,  // (C)
    float* __restrict__ proj)          // (B*T, C)
{
    __shared__ float s_t[RPB_][TE_];
    const int row0 = blockIdx.x * RPB_;
    for (int i = threadIdx.x; i < RPB_ * TE_; i += 256)
        s_t[i / TE_][i % TE_] = temb[(size_t)row0 * TE_ + i];
    __syncthreads();

    #pragma unroll
    for (int cc = 0; cc < 2; ++cc) {
        const int c = threadIdx.x + cc * 256;
        float acc[RPB_];
        #pragma unroll
        for (int r = 0; r < RPB_; ++r) acc[r] = b_time[c];
        for (int te = 0; te < TE_; ++te) {
            const float wv = w_time[te * C_ + c];
            #pragma unroll
            for (int r = 0; r < RPB_; ++r)
                acc[r] = fmaf(s_t[r][te], wv, acc[r]);
        }
        #pragma unroll
        for (int r = 0; r < RPB_; ++r)
            proj[(size_t)(row0 + r) * C_ + c] = acc[r];
    }
}

// ---------------------------------------------------------------------------
// Kernel 2: Bt[n, k] = (f16) w_out[k, n]  (transpose+convert, 512x512)
// ---------------------------------------------------------------------------
__global__ __launch_bounds__(256) void wout_t_kernel(
    const float* __restrict__ w_out,   // (C, C) row-major [k, n]
    _Float16* __restrict__ Bt)         // (C, C) row-major [n, k]
{
    const int n = blockIdx.x;
    for (int k = threadIdx.x; k < C_; k += 256)
        Bt[n * C_ + k] = (_Float16)w_out[k * C_ + n];
}

// ---------------------------------------------------------------------------
// Kernel 3: A[row, c] = (f16) silu(proj[row/S, c] + e0*wd0[c] + e1*wd1[c]
//                                 + e2*wd2[c] + b_dist[c]),  row = (b*T+t)*S+s
// ---------------------------------------------------------------------------
__global__ __launch_bounds__(256) void emb_silu_kernel(
    const int*   __restrict__ pd,      // (B*T*S)
    const float* __restrict__ proj,    // (B*T, C)
    const float* __restrict__ w_dist,  // (3, C)
    const float* __restrict__ b_dist,  // (C)
    _Float16* __restrict__ A)          // (B*T*S, C)
{
    const int row = blockIdx.x;
    const int d   = pd[row];
    const float df = (float)d;
    const float e0 = log1pf(fmaxf(df, 0.0f));
    const float e1 = log1pf(fmaxf(-df, 0.0f));
    const float e2 = (d == 0) ? 1.0f : 0.0f;
    const float* pr = proj + (row >> 7) * C_;   // S_ = 128
    for (int c = threadIdx.x; c < C_; c += 256) {
        float x = pr[c] + e0 * w_dist[c] + e1 * w_dist[C_ + c]
                        + e2 * w_dist[2 * C_ + c] + b_dist[c];
        float s = x / (1.0f + __expf(-x));      // silu
        A[(size_t)row * C_ + c] = (_Float16)s;
    }
}

// ---------------------------------------------------------------------------
// Kernel 4: R = A (32768 x 512) @ w_out (512 x 512) + b_out, stored f16.
// Wave tile 16(M) x 64(N); K double-buffered so loads for k+32 are in flight
// while WMMAs consume k (distinct register buffers -> partial s_wait_loadcnt).
// ---------------------------------------------------------------------------
__global__ __launch_bounds__(256) void rpe_gemm_kernel(
    const _Float16* __restrict__ A,    // (32768, 512)
    const _Float16* __restrict__ Bt,   // (512 n, 512 k)
    const float*    __restrict__ b_out,
    _Float16* __restrict__ R)          // (32768, 512)
{
    const int tid  = threadIdx.x;
    const int lane = tid & 31;
    const int wave = tid >> 5;
    const int ml   = lane & 15;   // M row (A/D) or N col (B/D)
    const int g    = lane >> 4;   // lane-group
    const int m0   = blockIdx.x * 16;
    const int n0   = wave * 64;

    const _Float16* arow = A + (size_t)(m0 + ml) * C_;
    const _Float16* brow0 = Bt + (size_t)(n0 + ml) * C_ + 16 * g;  // nt stride: 16*C_

    v8f acc[4] = {};
    V16 a[2];
    V16 b[2][4];

    auto load_k = [&](int buf, int k0) {
        // A 16x32 f16: lane-group g holds K=[k0+16g, +16)
        a[buf].h[0] = *(const v8h*)(arow + k0 + 16 * g);
        a[buf].h[1] = *(const v8h*)(arow + k0 + 16 * g + 8);
        #pragma unroll
        for (int nt = 0; nt < 4; ++nt) {
            // B 32x16: lane (n=ml, group g) holds K=[k0+16g, +16)
            const _Float16* p = brow0 + (size_t)nt * 16 * C_ + k0;
            b[buf][nt].h[0] = *(const v8h*)(p);
            b[buf][nt].h[1] = *(const v8h*)(p + 8);
        }
    };

    load_k(0, 0);
    #pragma unroll
    for (int k0 = 0; k0 < C_; k0 += 64) {
        load_k(1, k0 + 32);
        #pragma unroll
        for (int nt = 0; nt < 4; ++nt)
            acc[nt] = wmma_f16(a[0].v, b[0][nt].v, acc[nt]);
        if (k0 + 64 < C_) load_k(0, k0 + 64);
        #pragma unroll
        for (int nt = 0; nt < 4; ++nt)
            acc[nt] = wmma_f16(a[1].v, b[1][nt].v, acc[nt]);
    }

    // C/D 16x16 f32 layout: VGPR r, lane -> M = r + 8*g, N = ml
    #pragma unroll
    for (int nt = 0; nt < 4; ++nt) {
        const int n = n0 + nt * 16 + ml;
        const float bo = b_out[n];
        #pragma unroll
        for (int r = 0; r < 8; ++r) {
            const int m = m0 + r + 8 * g;
            R[(size_t)m * C_ + n] = (_Float16)(acc[nt][r] + bo);
        }
    }
}

// ---------------------------------------------------------------------------
// Kernel 5: out[b,d,h,t,s] = sum_f qk[b,d,h,t,f] * R[(b*T+t)*S+s, h*F+f]
// Per (b,h,t): (256 x 64) @ (64 x 128). Wave tile 32 x 64 (8 acc tiles).
// All operands (both K chunks) are loaded into distinct buffers before the
// 16 WMMAs so the load clause fully overlaps with the math.
// ---------------------------------------------------------------------------
__global__ __launch_bounds__(256) void einsum_kernel(
    const float*    __restrict__ qk,   // (B,D,H,T,F)
    const _Float16* __restrict__ R,    // (B*T*S, C)
    float* __restrict__ out)           // (B,D,H,T,S)
{
    const int tid  = threadIdx.x;
    const int lane = tid & 31;
    const int wave = tid >> 5;
    const int ml   = lane & 15;
    const int g    = lane >> 4;

    const int bi   = blockIdx.x;
    const int sblk = bi & 1;
    const int bht  = bi >> 1;
    const int t = bht & (T_ - 1);
    const int h = (bht >> 7) & (H_ - 1);
    const int b = bht >> 10;
    const int d0 = wave * 32;
    const int s0 = sblk * 64;

    v8f acc[2][4] = {};
    V16 a[2][2];                 // [kstep][mt]
    V16 bm[2][4];                // [kstep][nt]

    // ---- load + convert all A (qk fp32 -> f16) for both K chunks ----
    #pragma unroll
    for (int ks = 0; ks < 2; ++ks) {
        const int k0 = ks * 32;
        #pragma unroll
        for (int mt = 0; mt < 2; ++mt) {
            const int d = d0 + mt * 16 + ml;
            const float* qrow =
                qk + ((((size_t)b * D_ + d) * H_ + h) * T_ + t) * F_;
            #pragma unroll
            for (int c = 0; c < 2; ++c) {        // two 8-wide K chunks
                const float* p = qrow + k0 + c * 8 + 16 * g;
                v4f f0 = *(const v4f*)(p);
                v4f f1 = *(const v4f*)(p + 4);
                #pragma unroll
                for (int j = 0; j < 4; ++j) {
                    a[ks][mt].v[c * 8 + j]     = (_Float16)f0[j];
                    a[ks][mt].v[c * 8 + 4 + j] = (_Float16)f1[j];
                }
            }
        }
    }
    // ---- load all B tiles for both K chunks ----
    #pragma unroll
    for (int ks = 0; ks < 2; ++ks) {
        const int k0 = ks * 32;
        #pragma unroll
        for (int nt = 0; nt < 4; ++nt) {
            const int s = s0 + nt * 16 + ml;
            const _Float16* brow =
                R + ((size_t)(b * T_ + t) * S_ + s) * C_ + h * F_ + k0 + 16 * g;
            bm[ks][nt].h[0] = *(const v8h*)(brow);
            bm[ks][nt].h[1] = *(const v8h*)(brow + 8);
        }
    }
    // ---- 16 WMMAs ----
    #pragma unroll
    for (int ks = 0; ks < 2; ++ks)
        #pragma unroll
        for (int nt = 0; nt < 4; ++nt)
            #pragma unroll
            for (int mt = 0; mt < 2; ++mt)
                acc[mt][nt] = wmma_f16(a[ks][mt].v, bm[ks][nt].v, acc[mt][nt]);

    // ---- store (C layout: VGPR r, lane -> M = r + 8*g, N = ml) ----
    #pragma unroll
    for (int mt = 0; mt < 2; ++mt) {
        #pragma unroll
        for (int nt = 0; nt < 4; ++nt) {
            const int s = s0 + nt * 16 + ml;
            #pragma unroll
            for (int r = 0; r < 8; ++r) {
                const int d = d0 + mt * 16 + r + 8 * g;
                const size_t idx =
                    ((((size_t)b * D_ + d) * H_ + h) * T_ + t) * S_ + s;
                out[idx] = acc[mt][nt][r];
            }
        }
    }
}

// ---------------------------------------------------------------------------
extern "C" void kernel_launch(void* const* d_in, const int* in_sizes, int n_in,
                              void* d_out, int out_size, void* d_ws, size_t ws_size,
                              hipStream_t stream) {
    const float* qk     = (const float*)d_in[0];
    const float* temb   = (const float*)d_in[1];
    const int*   pd     = (const int*)  d_in[2];
    const float* w_dist = (const float*)d_in[3];
    const float* b_dist = (const float*)d_in[4];
    const float* w_time = (const float*)d_in[5];
    const float* b_time = (const float*)d_in[6];
    const float* w_out  = (const float*)d_in[7];
    const float* b_out  = (const float*)d_in[8];
    float* out = (float*)d_out;

    // Workspace layout (65 MB total)
    char* ws = (char*)d_ws;
    float*    proj = (float*)ws;                                   // 512 KB
    _Float16* Af16 = (_Float16*)(ws + (512u << 10));               // 32 MB
    _Float16* Bt   = (_Float16*)(ws + (512u << 10) + (32u << 20)); // 512 KB
    _Float16* R16  = (_Float16*)(ws + (1u << 20) + (32u << 20));   // 32 MB

    const int ROWS = B_ * T_ * S_;  // 32768

    temb_proj_kernel<<<(B_ * T_) / RPB_, 256, 0, stream>>>(temb, w_time, b_time, proj);
    wout_t_kernel   <<<C_,      256, 0, stream>>>(w_out, Bt);
    emb_silu_kernel <<<ROWS,    256, 0, stream>>>(pd, proj, w_dist, b_dist, Af16);
    rpe_gemm_kernel <<<ROWS / 16, 256, 0, stream>>>(Af16, Bt, b_out, R16);
    einsum_kernel   <<<B_ * H_ * T_ * 2, 256, 0, stream>>>(qk, R16, out);
}
